// UPFD_Net_20194936226508
// MI455X (gfx1250) — matrix operand order repaired and checked by reading, not compiled, not run on hardware
//
#include <hip/hip_runtime.h>
#include <hip/hip_bf16.h>

typedef __attribute__((ext_vector_type(16))) _Float16 v16h;
typedef __attribute__((ext_vector_type(8)))  _Float16 v8h;
typedef __attribute__((ext_vector_type(8)))  float    v8f;

#define IN_CH 768
#define OC 128
#define NGRAPH 128
#define NCLASS 2
#define BPAD 776                     // LDS column stride in halves (768 + 8 pad, kills bank conflicts)
#define B_LDS_BYTES (OC * BPAD * 2)  // 198,656 B < 320 KB per WGP

// ---------------- init workspace (re-run every call; ws is not re-poisoned) ---
__global__ void init_ws_kernel(float* __restrict__ acc, int* __restrict__ deg,
                               unsigned* __restrict__ pool, int* __restrict__ root,
                               long long nacc, int N) {
  long long i = (long long)blockIdx.x * blockDim.x + threadIdx.x;
  if (i < nacc) acc[i] = 0.0f;
  if (i < N) deg[i] = 0;
  if (i < (long long)NGRAPH * OC) pool[i] = 0u;
  if (i < NGRAPH) root[i] = 0;
}

// ---------------- degree: each undirected edge contributes to both endpoints --
__global__ void deg_kernel(const int* __restrict__ src, const int* __restrict__ dst,
                           int* __restrict__ deg, int E) {
  int e = blockIdx.x * blockDim.x + threadIdx.x;
  if (e >= E) return;
  atomicAdd(&deg[src[e]], 1);
  atomicAdd(&deg[dst[e]], 1);
}

// dinv = rsqrt(deg + 1)   (+1 = self loop)
__global__ void dinv_kernel(const int* __restrict__ deg, float* __restrict__ dinv, int N) {
  int v = blockIdx.x * blockDim.x + threadIdx.x;
  if (v >= N) return;
  dinv[v] = rsqrtf((float)(deg[v] + 1));
}

// root[g] = first node of graph g (batch is sorted)
__global__ void root_kernel(const int* __restrict__ batch, int* __restrict__ root, int N) {
  int i = blockIdx.x * blockDim.x + threadIdx.x;
  if (i >= N) return;
  int g = batch[i];
  if (i == 0 || batch[i - 1] != g) root[g] = i;
}

// W_conv [K=768][N=128] f32  ->  Wt [N][K] f16 (transposed, WMMA-B friendly)
__global__ void wconv_f16_kernel(const float* __restrict__ W, _Float16* __restrict__ Wt) {
  int idx = blockIdx.x * blockDim.x + threadIdx.x;
  if (idx >= IN_CH * OC) return;
  int k = idx / OC, n = idx % OC;
  Wt[(size_t)n * IN_CH + k] = (_Float16)W[idx];
}

// ---------------- big GEMM: h' = dinv[row] * (x @ W_conv), f16 WMMA, f32 acc --
// B (the whole 128x768 f16 weight) is staged into LDS once per block: all 8
// waves consume identical B tiles, so B reads become short-latency ds_loads
// instead of per-tile L2 round trips.
extern __shared__ _Float16 bs[];   // OC * BPAD halves

__global__ __launch_bounds__(256)
void gcn_gemm_wmma(const float* __restrict__ x, const _Float16* __restrict__ Wt,
                   const float* __restrict__ dinv, float* __restrict__ h, int N) {
  const int tid     = threadIdx.x;
  const int wave    = tid >> 5;           // 0..7, each wave: 16 rows x 128 cols
  const int lane    = tid & 31;
  const int halfSel = lane >> 4;          // 0: lanes 0-15, 1: lanes 16-31
  const int laneM   = lane & 15;
  const int m0      = blockIdx.x * 128 + wave * 16;

  // ---- stage Wt (f16, transposed) into padded LDS, 16B chunks ----
  for (int e = tid * 8; e < OC * IN_CH; e += 256 * 8) {
    int col = e / IN_CH;
    int k   = e - col * IN_CH;
    *(uint4*)(bs + col * BPAD + k) = *(const uint4*)(Wt + (size_t)col * IN_CH + k);
  }
  __syncthreads();

  // A-operand row for this lane (lanes 0-15 and 16-31 both map M = laneM)
  int row  = m0 + laneM;
  int rowc = row < N ? row : N - 1;       // clamp: EXEC must stay all-1 for WMMA
  const float* xrow = x + (size_t)rowc * IN_CH;
  // per-lane base into LDS weights: column = laneM, K-octet by halfSel (table order)
  const _Float16* lb = bs + laneM * BPAD + 8 * halfSel;

  v8f acc[8];
#pragma unroll
  for (int t = 0; t < 8; ++t)
#pragma unroll
    for (int j = 0; j < 8; ++j) acc[t][j] = 0.0f;

  for (int k0 = 0; k0 < IN_CH; k0 += 32) {
    // ---- A loads (the real HBM stream) + f32->f16 conversion ----
    // A 16x32 f16 layout: lanes<16 halves = {K0..7, K16..23}; lanes>=16 = {K8..15, K24..31}
    const float4* p0 = (const float4*)(xrow + k0 + 8 * halfSel);
    const float4* p1 = (const float4*)(xrow + k0 + 16 + 8 * halfSel);
    float4 q0 = p0[0], q1 = p0[1];
    float4 q2 = p1[0], q3 = p1[1];
    v16h a;
    a[0]  = (_Float16)q0.x; a[1]  = (_Float16)q0.y; a[2]  = (_Float16)q0.z; a[3]  = (_Float16)q0.w;
    a[4]  = (_Float16)q1.x; a[5]  = (_Float16)q1.y; a[6]  = (_Float16)q1.z; a[7]  = (_Float16)q1.w;
    a[8]  = (_Float16)q2.x; a[9]  = (_Float16)q2.y; a[10] = (_Float16)q2.z; a[11] = (_Float16)q2.w;
    a[12] = (_Float16)q3.x; a[13] = (_Float16)q3.y; a[14] = (_Float16)q3.z; a[15] = (_Float16)q3.w;

    // ---- WMMA chain; B tiles come from LDS (ds_load_b128 x2, conflict-free) ----
#pragma unroll
    for (int t = 0; t < 8; ++t) {
      const v8h* bp = (const v8h*)(lb + t * 16 * BPAD + k0);
      v8h b0 = bp[0], b1 = bp[2];
      v16h b;
#pragma unroll
      for (int i = 0; i < 8; ++i) { b[i] = b0[i]; b[8 + i] = b1[i]; }
      acc[t] = __builtin_amdgcn_wmma_f32_16x16x32_f16(
          false, a, false, b, (short)0, acc[t], false, false);
    }
  }

  // D layout: element j -> row m0 + j + 8*halfSel, col = t*16 + laneM
  float dv[8];
#pragma unroll
  for (int j = 0; j < 8; ++j) {
    int r = m0 + 8 * halfSel + j;
    dv[j] = (r < N) ? dinv[r] : 0.0f;
  }
#pragma unroll
  for (int t = 0; t < 8; ++t) {
    int col = t * 16 + laneM;
#pragma unroll
    for (int j = 0; j < 8; ++j) {
      int r = m0 + 8 * halfSel + j;
      if (r < N) h[(size_t)r * OC + col] = acc[t][j] * dv[j];
    }
  }
}

// ---------------- edge scatter: acc[d] += h'[s], acc[s] += h'[d]  (L2-resident)
__global__ void scatter_kernel(const int* __restrict__ src, const int* __restrict__ dst,
                               const float* __restrict__ h, float* __restrict__ acc, int E) {
  long long gid = (long long)blockIdx.x * blockDim.x + threadIdx.x;
  long long e = gid >> 5;                 // one wave32 per edge, 4 channels per lane
  if (e >= E) return;
  int lane = (int)(gid & 31);
  int s = src[e], d = dst[e];
  int c = lane * 4;
  const float4 hs = *(const float4*)(h + (size_t)s * OC + c);
  const float4 hd = *(const float4*)(h + (size_t)d * OC + c);
  float* ad = acc + (size_t)d * OC + c;
  float* as = acc + (size_t)s * OC + c;
  atomicAdd(ad + 0, hs.x); atomicAdd(ad + 1, hs.y);
  atomicAdd(ad + 2, hs.z); atomicAdd(ad + 3, hs.w);
  atomicAdd(as + 0, hd.x); atomicAdd(as + 1, hd.y);
  atomicAdd(as + 2, hd.z); atomicAdd(as + 3, hd.w);
}

// out = relu(dinv[v]*(acc + self-loop h') + b) ; max-pool per graph via u32 atomicMax
__global__ void finalize_kernel(const float* __restrict__ h, const float* __restrict__ acc,
                                const float* __restrict__ dinv, const float* __restrict__ bconv,
                                const int* __restrict__ batch, unsigned* __restrict__ pool, int N) {
  long long i = (long long)blockIdx.x * blockDim.x + threadIdx.x;
  if (i >= (long long)N * OC) return;
  int v = (int)(i >> 7);
  int c = (int)(i & (OC - 1));
  float val = dinv[v] * (acc[i] + h[i]) + bconv[c];
  val = fmaxf(val, 0.0f);   // >= 0 -> f32 bit pattern is order-preserving as u32
  atomicMax(&pool[(size_t)batch[v] * OC + c], __float_as_uint(val));
}

// news = relu(x[root] @ W_lin0 + b_lin0), one block per graph, f32
__global__ __launch_bounds__(128)
void news_kernel(const float* __restrict__ x, const float* __restrict__ W0,
                 const float* __restrict__ b0, const int* __restrict__ root,
                 float* __restrict__ news) {
  __shared__ float xr[IN_CH];
  int g = blockIdx.x, c = threadIdx.x;
  int r = root[g];
  for (int i = c; i < IN_CH; i += 128) xr[i] = x[(size_t)r * IN_CH + i];
  __syncthreads();
  float s = b0[c];
  for (int k = 0; k < IN_CH; ++k) s += xr[k] * W0[(size_t)k * OC + c];
  news[(size_t)g * OC + c] = fmaxf(s, 0.0f);
}

// lin1 -> relu -> lin2 -> log_softmax, one block per graph
__global__ __launch_bounds__(128)
void head_kernel(const float* __restrict__ news, const unsigned* __restrict__ pool,
                 const float* __restrict__ W1, const float* __restrict__ b1,
                 const float* __restrict__ W2, const float* __restrict__ b2,
                 float* __restrict__ out) {
  __shared__ float feat[2 * OC];
  __shared__ float hid[OC];
  __shared__ float lg[NCLASS];
  int g = blockIdx.x, c = threadIdx.x;
  feat[c]      = news[(size_t)g * OC + c];
  feat[OC + c] = __uint_as_float(pool[(size_t)g * OC + c]);
  __syncthreads();
  float s = b1[c];
  for (int j = 0; j < 2 * OC; ++j) s += feat[j] * W1[(size_t)j * OC + c];
  hid[c] = fmaxf(s, 0.0f);
  __syncthreads();
  if (c < NCLASS) {
    float l = b2[c];
    for (int k = 0; k < OC; ++k) l += hid[k] * W2[(size_t)k * NCLASS + c];
    lg[c] = l;
  }
  __syncthreads();
  if (c < NCLASS) {
    float m = fmaxf(lg[0], lg[1]);
    float lse = m + logf(expf(lg[0] - m) + expf(lg[1] - m));
    out[(size_t)g * NCLASS + c] = lg[c] - lse;
  }
}

extern "C" void kernel_launch(void* const* d_in, const int* in_sizes, int n_in,
                              void* d_out, int out_size, void* d_ws, size_t ws_size,
                              hipStream_t stream) {
  const float* x     = (const float*)d_in[0];
  const int*   ei    = (const int*)d_in[1];
  const int*   batch = (const int*)d_in[2];
  const float* Wc    = (const float*)d_in[4];
  const float* bc    = (const float*)d_in[5];
  const float* W0    = (const float*)d_in[6];
  const float* b0    = (const float*)d_in[7];
  const float* W1    = (const float*)d_in[8];
  const float* b1    = (const float*)d_in[9];
  const float* W2    = (const float*)d_in[10];
  const float* b2    = (const float*)d_in[11];
  float* out = (float*)d_out;

  const int N = in_sizes[0] / IN_CH;       // 100000
  const int E = in_sizes[1] / 2;           // 1600000
  const int* src = ei;
  const int* dst = ei + E;

  // workspace carve-up (all 256B aligned); total ~104 MB
  char* p = (char*)d_ws;
  auto carve = [&](size_t bytes) -> char* {
    char* r = p; p += (bytes + 255) & ~(size_t)255; return r;
  };
  float*    acc  = (float*)   carve((size_t)N * OC * sizeof(float));
  float*    h    = (float*)   carve((size_t)N * OC * sizeof(float));
  int*      deg  = (int*)     carve((size_t)N * sizeof(int));
  float*    dinv = (float*)   carve((size_t)N * sizeof(float));
  unsigned* pool = (unsigned*)carve((size_t)NGRAPH * OC * sizeof(unsigned));
  int*      root = (int*)     carve((size_t)NGRAPH * sizeof(int));
  float*    news = (float*)   carve((size_t)NGRAPH * OC * sizeof(float));
  _Float16* Wt   = (_Float16*)carve((size_t)IN_CH * OC * sizeof(_Float16));

  const int TB = 256;
  const long long nacc = (long long)N * OC;

  init_ws_kernel<<<(unsigned)((nacc + TB - 1) / TB), TB, 0, stream>>>(acc, deg, pool, root, nacc, N);
  deg_kernel<<<(E + TB - 1) / TB, TB, 0, stream>>>(src, dst, deg, E);
  dinv_kernel<<<(N + TB - 1) / TB, TB, 0, stream>>>(deg, dinv, N);
  root_kernel<<<(N + TB - 1) / TB, TB, 0, stream>>>(batch, root, N);
  wconv_f16_kernel<<<(IN_CH * OC + TB - 1) / TB, TB, 0, stream>>>(Wc, Wt);

  gcn_gemm_wmma<<<(N + 127) / 128, 256, B_LDS_BYTES, stream>>>(x, Wt, dinv, h, N);

  long long sthreads = (long long)E * 32;
  scatter_kernel<<<(unsigned)((sthreads + TB - 1) / TB), TB, 0, stream>>>(src, dst, h, acc, E);
  finalize_kernel<<<(unsigned)((nacc + TB - 1) / TB), TB, 0, stream>>>(h, acc, dinv, bc, batch, pool, N);

  news_kernel<<<NGRAPH, 128, 0, stream>>>(x, W0, b0, root, news);
  head_kernel<<<NGRAPH, 128, 0, stream>>>(news, pool, W1, b1, W2, b2, out);
}
